// Anti_Alias_Filter_36146444763206
// MI455X (gfx1250) — matrix operand order repaired and checked by reading, not compiled, not run on hardware
//
#include <hip/hip_runtime.h>
#include <hip/hip_bf16.h>

typedef __attribute__((ext_vector_type(16))) __bf16 v16bf;
typedef __attribute__((ext_vector_type(8)))  __bf16 v8bf;
typedef __attribute__((ext_vector_type(8)))  float  v8f;
typedef __attribute__((ext_vector_type(4))) unsigned int u32x4;
typedef __attribute__((ext_vector_type(8))) int i32x8;
typedef __attribute__((ext_vector_type(4))) int i32x4;

#define GK    72     // G*K*K output channels of the conv
#define CIN   256
#define CG    32     // channels per group
#define TH    4      // tile height (pixels)
#define TW    32     // tile width (pixels)
#define PH    6      // TH + 2 (halo)
#define PW    34     // TW + 2 (halo)
#define CPAD  280    // padded channel stride in halfs (bank-friendly: 140 dwords)
#define NPIX  128    // TH*TW pixels per workgroup
#define HW    128
#define WELEM (GK * CG * 9)   // 20736 conv weights

// ---- LDS layout (bytes) ----
#define OFF_PATCH 0
#define SZ_PATCH  (PH * PW * CPAD * 2)            // 114240
#define OFF_WFRAG (OFF_PATCH + SZ_PATCH)
#define SZ_WFRAG  (GK * 32 * 16 * 2)              // 73728: A-fragments, lane-row contiguous
#define OFF_SIGMA (OFF_WFRAG + SZ_WFRAG)
#define SZ_SIGMA  (GK * NPIX * 4)                 // 36864
#define OFF_SCALE (OFF_SIGMA + SZ_SIGMA)
#define OFF_SHIFT (OFF_SCALE + GK * 4)
#define OFF_WSTAGE (OFF_SHIFT + GK * 4)           // raw f32 weights staged by TDM
#define SZ_WSTAGE  (WELEM * 4)                    // 82944
#define SMEM_BYTES (OFF_WSTAGE + SZ_WSTAGE)       // 308352 < 320K

#if __has_builtin(__builtin_amdgcn_tensor_load_to_lds)
#define USE_TDM 1
#else
#define USE_TDM 0
#endif

__device__ __forceinline__ unsigned short f2bf(float f) {
    unsigned int u = __float_as_uint(f);
    u += 0x7FFFu + ((u >> 16) & 1u);              // round-to-nearest-even
    return (unsigned short)(u >> 16);
}
__device__ __forceinline__ float bf2f(unsigned short h) {
    return __uint_as_float(((unsigned int)h) << 16);
}
__device__ __forceinline__ int reflect1(int i, int n) {
    if (i < 0) i = -i;
    if (i >= n) i = 2 * n - 2 - i;
    return i;
}
__device__ __forceinline__ v16bf ld_frag16(const unsigned short* p) {
    v8bf lo = *(const v8bf*)(p);
    v8bf hi = *(const v8bf*)(p + 8);
    return __builtin_shufflevector(lo, hi, 0,1,2,3,4,5,6,7,8,9,10,11,12,13,14,15);
}

__global__ __launch_bounds__(256, 1)
void Anti_Alias_Filter_kernel(const float* __restrict__ x,
                              const float* __restrict__ w,
                              const float* __restrict__ gamma,
                              const float* __restrict__ beta,
                              const float* __restrict__ rmean,
                              const float* __restrict__ rvar,
                              float* __restrict__ out)
{
    extern __shared__ char smem[];
    unsigned short* patchU = (unsigned short*)(smem + OFF_PATCH);
    unsigned short* wfragU = (unsigned short*)(smem + OFF_WFRAG);
    float* sigmaS = (float*)(smem + OFF_SIGMA);
    float* scaleS = (float*)(smem + OFF_SCALE);
    float* shiftS = (float*)(smem + OFF_SHIFT);

    const int tid  = threadIdx.x;
    const int wave = tid >> 5;
    const int lane = tid & 31;
    const int n  = blockIdx.z;
    const int y0 = blockIdx.y * TH;
    const int x0 = blockIdx.x * TW;

    // --------- Phase 0: kick off TDM DMA of conv weights (global -> LDS) ---------
    // One tensor_load_to_lds moves all 82944B of conv_w; overlaps with Phase 1a.
#if USE_TDM
    if (tid == 0) {
        const unsigned int      ldsA = (unsigned int)(size_t)(smem + OFF_WSTAGE);
        const unsigned long long ga  = (unsigned long long)(size_t)w;
        u32x4 g0;
        g0[0] = 1u;                                            // count=1, user D#
        g0[1] = ldsA;                                          // lds_addr
        g0[2] = (unsigned int)(ga & 0xFFFFFFFFull);            // global_addr[31:0]
        g0[3] = (unsigned int)((ga >> 32) & 0x1FFFFFFull)      // global_addr[56:32]
              | (2u << 30);                                    // type=2 ("image")
        i32x8 g1;
        g1[0] = (int)(2u << 16);                               // wg_mask=0, data_size=4B
        g1[1] = (int)(((unsigned)WELEM & 0xFFFFu) << 16);      // tensor_dim0[15:0]
        g1[2] = (int)(((unsigned)WELEM >> 16) | (1u << 16));   // tensor_dim0[31:16] | tensor_dim1=1
        g1[3] = (int)(((unsigned)WELEM & 0xFFFFu) << 16);      // tile_dim0 = WELEM
        g1[4] = 1;                                             // tile_dim1=1, tile_dim2=0
        g1[5] = (int)WELEM;                                    // tensor_dim0_stride[31:0]
        g1[6] = 0;
        g1[7] = 0;
        i32x4 z4 = {0, 0, 0, 0};                               // <=2D: groups 2/3 unused
#if __has_include(<hip/amd_detail/amd_gfx1250_TDM.h>)
        i32x8 z8 = {0, 0, 0, 0, 0, 0, 0, 0};
        __builtin_amdgcn_tensor_load_to_lds(g0, g1, z4, z4, z8, 0);
#else
        __builtin_amdgcn_tensor_load_to_lds(g0, g1, z4, z4, 0);
#endif
    }
#endif

    // ---------------- Phase 1a: x patch -> LDS bf16 (reflect pad) ----------------
    {
        const int gxa = reflect1(x0 - 1 + lane, HW);                // cols 0..31
        const int gxb = (lane < PW - 32) ? reflect1(x0 + 31 + lane, HW) : 0; // cols 32..33
        for (int job = wave; job < CIN * PH; job += 8) {
            const int c = job / PH;
            const int r = job % PH;
            const int gy = reflect1(y0 - 1 + r, HW);
            const float* src = x + (((size_t)n * CIN + c) * HW + gy) * HW;
            __builtin_prefetch(src + HW * HW, 0, 1);                // next channel, same row
            unsigned short* dst = patchU + (r * PW) * CPAD + c;
            dst[lane * CPAD] = f2bf(src[gxa]);
            if (lane < PW - 32) dst[(32 + lane) * CPAD] = f2bf(src[gxb]);
        }
    }

#if USE_TDM
#if __has_builtin(__builtin_amdgcn_s_wait_tensorcnt)
    if (tid == 0) __builtin_amdgcn_s_wait_tensorcnt(0);        // DMA landed
#endif
    __syncthreads();                                           // publish staged weights
    const float* wsrc = (const float*)(smem + OFF_WSTAGE);
#else
    const float* wsrc = w;
#endif

    // ------------- Phase 1b: weights -> WMMA A-fragment layout in LDS -------------
    // A (16x32 bf16): lane l row M=l%16; half h -> K = (l<16?0:16) + h
    for (int i = 0; i < 9; ++i) {
        const int fr = tid + 256 * i;            // fragment-lane index, 2304 total
        const int f  = fr >> 5;                  // fragment = g*9 + tap
        const int fl = fr & 31;
        const int g = f / 9, tap = f % 9;
        const int o  = fl & 15;
        const int kb = (fl < 16) ? 0 : 16;
        unsigned short* dst = wfragU + fr * 16;
        #pragma unroll
        for (int h = 0; h < 16; ++h) {
            const int ci = kb + h;
            float v = 0.f;
            if (o < 9) v = wsrc[((g * 9 + o) * CG + ci) * 9 + tap];
            dst[h] = f2bf(v);
        }
    }

    // ------------------------------ Phase 1c: BN ---------------------------------
    if (tid < GK) {
        const float s = gamma[tid] * rsqrtf(rvar[tid] + 1e-5f);
        scaleS[tid] = s;
        shiftS[tid] = beta[tid] - rmean[tid] * s;
    }
    __syncthreads();

    // --------------- Phase 2: WMMA conv + BN + softmax (per wave) ----------------
    // wave owns 16 pixels: row py = wave>>1, cols pxb..pxb+15
    const int py  = wave >> 1;
    const int pxb = (wave & 1) * 16;
    const int N   = lane & 15;
    const int px  = pxb + N;
    const int kbB = (lane < 16) ? 0 : 16;        // B (32x16): lane col N; K = kbB + h

    v8f acc[8];
    #pragma unroll
    for (int g = 0; g < 8; ++g) {
        v8f cacc = {};
        const unsigned short* afrag = wfragU + ((g * 9) * 32 + lane) * 16;
        #pragma unroll
        for (int t = 0; t < 9; ++t) {
            const int dy = t / 3, dx = t % 3;
            v16bf a = *(const v16bf*)(afrag + t * 32 * 16);   // 32B aligned
            v16bf b = ld_frag16(patchU + ((py + dy) * PW + (px + dx)) * CPAD
                                        + g * CG + kbB);
            cacc = __builtin_amdgcn_wmma_f32_16x16x32_bf16(
                       false, a, false, b, (short)0, cacc, false, false);
        }
        acc[g] = cacc;
    }

    // C/D layout: lanes 0-15 hold M=0..7 (v0..v7), lanes 16-31 hold M=8..15.
    // Valid channels per group: M=0..8 (rows 9..15 are zero-weight padding, excluded).
    float mx = -3.4e38f;
    if (lane < 16) {
        #pragma unroll
        for (int g = 0; g < 8; ++g)
            #pragma unroll
            for (int m = 0; m < 8; ++m) {
                const float v = acc[g][m] * scaleS[g * 9 + m] + shiftS[g * 9 + m];
                acc[g][m] = v; mx = fmaxf(mx, v);
            }
    } else {
        #pragma unroll
        for (int g = 0; g < 8; ++g) {
            const float v = acc[g][0] * scaleS[g * 9 + 8] + shiftS[g * 9 + 8];
            acc[g][0] = v; mx = fmaxf(mx, v);
        }
    }
    mx = fmaxf(mx, __shfl_xor(mx, 16, 32));

    float sum = 0.f;
    if (lane < 16) {
        #pragma unroll
        for (int g = 0; g < 8; ++g)
            #pragma unroll
            for (int m = 0; m < 8; ++m) {
                const float e = __expf(acc[g][m] - mx);
                acc[g][m] = e; sum += e;
            }
    } else {
        #pragma unroll
        for (int g = 0; g < 8; ++g) {
            const float e = __expf(acc[g][0] - mx);
            acc[g][0] = e; sum += e;
        }
    }
    sum += __shfl_xor(sum, 16, 32);
    const float inv = 1.f / sum;

    const int p = wave * 16 + N;                 // pixel index within tile
    if (lane < 16) {
        #pragma unroll
        for (int g = 0; g < 8; ++g)
            #pragma unroll
            for (int m = 0; m < 8; ++m)
                sigmaS[(g * 9 + m) * NPIX + p] = acc[g][m] * inv;
    } else {
        #pragma unroll
        for (int g = 0; g < 8; ++g)
            sigmaS[(g * 9 + 8) * NPIX + p] = acc[g][0] * inv;
    }
    __syncthreads();

    // -------- Phase 3: 9-tap weighted sum; lanes = px -> coalesced stores --------
    for (int k = 0; k < 32; ++k) {
        const int ch = wave + 8 * k;
        const int g  = ch >> 5;
        const float* sg = sigmaS + (g * 9) * NPIX;
        float* orow = out + (((size_t)n * CIN + ch) * HW + y0) * HW + x0;
        #pragma unroll
        for (int r = 0; r < TH; ++r) {
            float o = 0.f;
            #pragma unroll
            for (int t = 0; t < 9; ++t) {
                const int dy = t / 3, dx = t % 3;
                const float s  = sg[t * NPIX + r * TW + lane];
                const float pv = bf2f(patchU[((r + dy) * PW + (lane + dx)) * CPAD + ch]);
                o = fmaf(s, pv, o);
            }
            orow[r * HW + lane] = o;             // 32 lanes * 4B = 128B coalesced
        }
    }
}

extern "C" void kernel_launch(void* const* d_in, const int* in_sizes, int n_in,
                              void* d_out, int out_size, void* d_ws, size_t ws_size,
                              hipStream_t stream) {
    const float* x     = (const float*)d_in[0];
    const float* w     = (const float*)d_in[1];
    const float* gamma = (const float*)d_in[2];
    const float* beta  = (const float*)d_in[3];
    const float* rmean = (const float*)d_in[4];
    const float* rvar  = (const float*)d_in[5];
    float* out = (float*)d_out;

    dim3 grid(HW / TW, HW / TH, 16);   // (4, 32, 16) = 2048 workgroups
    hipLaunchKernelGGL(Anti_Alias_Filter_kernel, grid, dim3(256), SMEM_BYTES, stream,
                       x, w, gamma, beta, rmean, rvar, out);
}
